// DynamicConceptGraphBuilder_21612275433819
// MI455X (gfx1250) — compile-verified, matrix operand order checked by reading, not run on hardware
//
#include <hip/hip_runtime.h>
#include <hip/hip_bf16.h>

#define NUM_C  8192
#define DIM    128
#define K_KEEP 17                 // TOP_K + 1 (self included)
#define WAVES  4
#define COLS_PER_WAVE (NUM_C / WAVES)   // 2048 -> 128 blocks of 16 (even)

typedef _Float16 half8_t  __attribute__((ext_vector_type(8)));
typedef _Float16 v16h_t   __attribute__((ext_vector_type(16)));
typedef float    v8f_t    __attribute__((ext_vector_type(8)));

// ---------------------------------------------------------------------------
// Kernel 1: cosine-normalize rows, f32 -> f16 (WMMA input precision).
// One wave32 per row; each lane owns 4 contiguous floats (32*4 = 128 = DIM).
// ---------------------------------------------------------------------------
__global__ __launch_bounds__(256) void normalize_f16_kernel(
    const float* __restrict__ mv, _Float16* __restrict__ normed) {
  const int row  = blockIdx.x * 8 + (threadIdx.x >> 5);
  const int lane = threadIdx.x & 31;

  const float4 v = reinterpret_cast<const float4*>(mv + (size_t)row * DIM)[lane];
  float ss = v.x * v.x + v.y * v.y + v.z * v.z + v.w * v.w;
  #pragma unroll
  for (int off = 16; off > 0; off >>= 1) ss += __shfl_xor(ss, off, 32);

  const float inv = 1.0f / fmaxf(sqrtf(ss), 1e-6f);

  _Float16* dst = normed + (size_t)row * DIM + lane * 4;
  dst[0] = (_Float16)(v.x * inv);
  dst[1] = (_Float16)(v.y * inv);
  dst[2] = (_Float16)(v.z * inv);
  dst[3] = (_Float16)(v.w * inv);
}

// One v16h fragment of a 16x32 (MxK) f16 tile whose rows are consecutive rows
// of `normed`. Valid for both A and B operands (Gram symmetry, M<->N).
// lanes 0-15: K {kb..kb+7, kb+16..kb+23}; lanes 16-31: shifted by +8.
__device__ __forceinline__ v16h_t load_frag(const _Float16* __restrict__ normed,
                                            int row0, int mN, int kbase) {
  const _Float16* base = normed + (size_t)(row0 + mN) * DIM + kbase;
  const half8_t lo = *reinterpret_cast<const half8_t*>(base);
  const half8_t hi = *reinterpret_cast<const half8_t*>(base + 16);
  v16h_t f;
  #pragma unroll
  for (int i = 0; i < 8; ++i) { f[i] = lo[i]; f[8 + i] = hi[i]; }
  return f;
}

// ---------------------------------------------------------------------------
// Kernel 2: streaming Gram + per-row top-17.
// 4 waves per block share one 16-row band; wave w scans columns
// [w*2048, (w+1)*2048). Ping-pong B fragments (2x-unrolled loop, no register
// copies) hide load latency behind the WMMA chain. Per 16-col block, a
// branchless fmax reduction gives a row-max early-out; only competitive
// blocks take the insert path. Wave 0 merges the 4 partial top-17 lists,
// sorts by column index, writes [rows | cols | weights].
// ---------------------------------------------------------------------------
__global__ __launch_bounds__(128) void gram_topk_kernel(
    const _Float16* __restrict__ normed, float* __restrict__ out) {
  __shared__ __align__(16) float tile[WAVES][16][20]; // 20: rows stay 16B-aligned
  __shared__ float lv[WAVES][16][K_KEEP];
  __shared__ int   li[WAVES][16][K_KEEP];
  __shared__ float fv[16][K_KEEP];
  __shared__ int   fi[16][K_KEEP];

  const int tid  = threadIdx.x;
  const int w    = tid >> 5;            // wave id 0..3
  const int lane = tid & 31;
  const int r0   = blockIdx.x * 16;     // band's first row
  const int mN   = lane & 15;
  const int hiG  = lane >> 4;
  const int koff = hiG * 8;

  // A fragments (identical across the 4 waves), resident for the whole loop.
  v16h_t afrag[4];
  #pragma unroll
  for (int q = 0; q < 4; ++q)
    afrag[q] = load_frag(normed, r0, mN, q * 32 + koff);

  const int nStart = w * COLS_PER_WAVE;
  const int nEnd   = nStart + COLS_PER_WAVE;

  float curMin = -3.0e38f;
  int   minPos = 0;
  if (lane < 16) {
    #pragma unroll
    for (int t = 0; t < K_KEEP; ++t) { lv[w][lane][t] = -3.0e38f; li[w][lane][t] = 0; }
  }

  auto load_b = [&](v16h_t (&b)[4], int rowStart) {
    #pragma unroll
    for (int q = 0; q < 4; ++q)
      b[q] = load_frag(normed, rowStart, mN, q * 32 + koff);
  };

  auto compute = [&](const v16h_t (&b)[4]) -> v8f_t {
    v8f_t c = {};
    #pragma unroll
    for (int q = 0; q < 4; ++q)
      c = __builtin_amdgcn_wmma_f32_16x16x32_f16(
              false, afrag[q], false, b[q], (short)0, c, false, false);
    return c;
  };

  auto process_block = [&](const v8f_t c, int n0) {
    // Spill C: lane<16 -> M=r, lane>=16 -> M=8+r ; N = lane&15.
    #pragma unroll
    for (int r = 0; r < 8; ++r) tile[w][hiG * 8 + r][mN] = c[r];

    // DS ops are in-order within a wave; fence so the row reads below cannot
    // be hoisted above the tile stores.
    asm volatile("s_wait_dscnt 0x0" ::: "memory");

    if (lane < 16) {
      // Row-owner lane pulls its 16 fresh similarities as 4x float4.
      const float4* trow = reinterpret_cast<const float4*>(&tile[w][lane][0]);
      const float4 t0 = trow[0], t1 = trow[1], t2 = trow[2], t3 = trow[3];

      const float bm =
          fmaxf(fmaxf(fmaxf(fmaxf(t0.x, t0.y), fmaxf(t0.z, t0.w)),
                      fmaxf(fmaxf(t1.x, t1.y), fmaxf(t1.z, t1.w))),
                fmaxf(fmaxf(fmaxf(t2.x, t2.y), fmaxf(t2.z, t2.w)),
                      fmaxf(fmaxf(t3.x, t3.y), fmaxf(t3.z, t3.w))));

      if (bm > curMin) {   // rare after warm-up: block contains a candidate
        const float vals[16] = {t0.x, t0.y, t0.z, t0.w, t1.x, t1.y, t1.z, t1.w,
                                t2.x, t2.y, t2.z, t2.w, t3.x, t3.y, t3.z, t3.w};
        #pragma unroll
        for (int j = 0; j < 16; ++j) {
          const float v = vals[j];
          if (v > curMin) {
            lv[w][lane][minPos] = v;
            li[w][lane][minPos] = n0 + j;
            float m = lv[w][lane][0]; int mp = 0;
            #pragma unroll
            for (int t = 1; t < K_KEEP; ++t) {
              const float x = lv[w][lane][t];
              if (x < m) { m = x; mp = t; }
            }
            curMin = m; minPos = mp;
          }
        }
      }
    }
  };

  // Ping-pong main loop: 2 blocks per trip, zero buffer copies.
  v16h_t bufA[4], bufB[4];
  load_b(bufA, nStart);
  for (int n0 = nStart; n0 < nEnd; n0 += 32) {
    load_b(bufB, n0 + 16);                    // in flight during block n0
    process_block(compute(bufA), n0);

    const int na = (n0 + 32 < nEnd) ? n0 + 32 : nStart;  // wrap: safe dummy
    load_b(bufA, na);                         // in flight during block n0+16
    process_block(compute(bufB), n0 + 16);
  }

  __syncthreads();   // all 4 partial lists are final

  // Wave 0, lanes 0-15: merge 4x17 candidates -> top-17, sort by col, emit.
  if (tid < 16) {
    const int m = tid;
    float cmin = -3.0e38f; int cpos = 0;
    #pragma unroll
    for (int t = 0; t < K_KEEP; ++t) { fv[m][t] = -3.0e38f; fi[m][t] = 0; }

    for (int w2 = 0; w2 < WAVES; ++w2) {
      #pragma unroll
      for (int t = 0; t < K_KEEP; ++t) {
        const float v  = lv[w2][m][t];
        const int   ix = li[w2][m][t];
        if (v > cmin) {
          fv[m][cpos] = v; fi[m][cpos] = ix;
          float mm = fv[m][0]; int mp = 0;
          #pragma unroll
          for (int t2 = 1; t2 < K_KEEP; ++t2) {
            const float x = fv[m][t2];
            if (x < mm) { mm = x; mp = t2; }
          }
          cmin = mm; cpos = mp;
        }
      }
    }

    // Insertion sort by column index ascending (matches jnp.sort of indices).
    for (int i = 1; i < K_KEEP; ++i) {
      const int   ki = fi[m][i];
      const float kv = fv[m][i];
      int j = i - 1;
      while (j >= 0 && fi[m][j] > ki) {
        fi[m][j + 1] = fi[m][j];
        fv[m][j + 1] = fv[m][j];
        --j;
      }
      fi[m][j + 1] = ki;
      fv[m][j + 1] = kv;
    }

    const int    row = r0 + m;
    const size_t E   = (size_t)NUM_C * K_KEEP;
    float* orow = out;                 // edge_index[0] (rows)
    float* ocol = out + E;             // edge_index[1] (cols)
    float* owgt = out + 2 * E;         // edge_weight
    #pragma unroll
    for (int t = 0; t < K_KEEP; ++t) {
      const size_t e = (size_t)row * K_KEEP + t;
      orow[e] = (float)row;
      ocol[e] = (float)fi[m][t];
      owgt[e] = fv[m][t];
    }
  }
}

extern "C" void kernel_launch(void* const* d_in, const int* in_sizes, int n_in,
                              void* d_out, int out_size, void* d_ws, size_t ws_size,
                              hipStream_t stream) {
  (void)in_sizes; (void)n_in; (void)out_size; (void)ws_size;

  const float* memory_value = (const float*)d_in[0];   // [8192, 128] f32
  float*       out          = (float*)d_out;           // rows|cols|weights (f32)
  _Float16*    normed       = (_Float16*)d_ws;         // [8192, 128] f16 (2 MB)

  normalize_f16_kernel<<<NUM_C / 8, 256, 0, stream>>>(memory_value, normed);
  gram_topk_kernel<<<NUM_C / 16, 128, 0, stream>>>(normed, out);
}
